// kernel_generated_0_62904091017341
// MI455X (gfx1250) — matrix operand order repaired and checked by reading, not compiled, not run on hardware
//
#include <hip/hip_runtime.h>

// ---------------------------------------------------------------------------
// out[o,i,m,n'] = y[o,i,(m-1)%56, l=1, n] + y[o,i,m, l=0, n],  n=(n'-1)%56
// y[.,i,m,l,.] = sum_{d<192} w[i,d/64,l,d%64] * Xc[o,(m-1)*64+d,.]
// Xc[o,c,n] = x[o, c%56, n, c/56]  (c = k*56 + l_x, zero outside [0,3584))
//
// f32 emulated as 3-term bf16 split on V_WMMA_F32_16X16X32_BF16.
// ---------------------------------------------------------------------------

typedef float        v8f   __attribute__((ext_vector_type(8)));
typedef unsigned int v8u   __attribute__((ext_vector_type(8)));
typedef __bf16       v16bf __attribute__((ext_vector_type(16)));

#define NBLK_N   4
#define NT       14                     // n columns per tile (B uses 16, 14 valid)
#define XPAD     64                     // zero-pad rows (front & back) of c-axis
#define XROWS    (XPAD + 3584 + XPAD)   // 3712 (even)
#define XPAIRS   (XROWS / 2)            // 1856 K-pairs
#define XD_DW    (XPAIRS * NT)          // 25,984 dwords per Xs half
#define WSTD     68                     // weight table dword stride (8*68 % 64 = 32 -> halves disjoint banks)
#define EPAIRS   96                     // 192 K / 2
#define WT_DW    (EPAIRS * WSTD)        // 6,528 dwords per weight table
#define SMEM_DW  (2 * XD_DW + 4 * WT_DW)  // 78,080 dw = 312,320 B (< 320 KB/WGP)

__device__ __forceinline__ void split_bf16(float v, __bf16& h, __bf16& l) {
  h = (__bf16)v;
  l = (__bf16)(v - (float)h);
}

// B 32x16 bf16 operand: VGPR v -> K-pair (v + 8*khalf), column = coln.
__device__ __forceinline__ v16bf ldB(const unsigned int* XD, int pairbase,
                                     int coln, int khalf) {
  v8u t;
  const int idx = (pairbase + 8 * khalf) * NT + coln;
#pragma unroll
  for (int v = 0; v < 8; ++v) t[v] = XD[idx + v * NT];
  return __builtin_bit_cast(v16bf, t);
}

// A 16x32 bf16 operand: VGPR v -> K-pair (v + 8*khalf).
__device__ __forceinline__ v16bf ldA(const unsigned int* WD, int pairbase,
                                     int aoff, int khalf) {
  v8u t;
  const int b0 = (pairbase + 8 * khalf) * WSTD + aoff;
#pragma unroll
  for (int v = 0; v < 4; ++v) t[v] = WD[b0 + v * WSTD];
  const int b1 = (pairbase + 4 + 8 * khalf) * WSTD + aoff;
#pragma unroll
  for (int v = 0; v < 4; ++v) t[4 + v] = WD[b1 + v * WSTD];
  return __builtin_bit_cast(v16bf, t);
}

__global__ __launch_bounds__(256) void shiftconv_wmma_bf16x3(
    const float* __restrict__ x,   // (1024, 56, 56, 64)
    const float* __restrict__ w,   // (64, 3, 2, 64)
    float* __restrict__ out) {     // (1024, 64, 56, 56)
  extern __shared__ unsigned int smem_u[];
  unsigned int* XDh  = smem_u;                 // Xs hi, [XPAIRS][NT] dwords
  unsigned int* XDl  = XDh + XD_DW;            // Xs lo
  unsigned int* WhiH = XDl + XD_DW;            // l=1 weights, hi half, [epair][i]
  unsigned int* WhiL = WhiH + WT_DW;
  unsigned int* WloH = WhiL + WT_DW;           // l=0 weights
  unsigned int* WloL = WloH + WT_DW;

  __bf16* XsH = (__bf16*)XDh;   // 16-bit elem index = dword*2 + (c&1)
  __bf16* XsL = (__bf16*)XDl;
  __bf16* WhH = (__bf16*)WhiH;
  __bf16* WhL = (__bf16*)WhiL;
  __bf16* WlH = (__bf16*)WloH;
  __bf16* WlL = (__bf16*)WloL;

  const int tid = threadIdx.x;
  const int o   = (int)blockIdx.x >> 2;
  const int n0  = ((int)blockIdx.x & 3) * NT;

  // ---- zero boundary pads (c in [0,64) and [3648,3712)) ------------------
  const int backBase = (XROWS - XPAD) * NT;    // 16-bit element index 51,072
  for (int idx = tid; idx < XPAD * NT; idx += 256) {
    XsH[idx] = (__bf16)0.f;            XsL[idx] = (__bf16)0.f;
    XsH[backBase + idx] = (__bf16)0.f; XsL[backBase + idx] = (__bf16)0.f;
  }

  // ---- stage x[o, :, n0:n0+14, :] permuted + bf16-split ------------------
  const float* xo = x + (size_t)o * 200704 + (size_t)n0 * 64;
  for (int g = tid; g < 56 * NT * 64; g += 256) {
    const int k  = g & 63;
    const int nn = (g >> 6) % NT;
    const int lx = g / (NT * 64);
    const float v = xo[lx * 3584 + nn * 64 + k];
    __bf16 h, l;
    split_bf16(v, h, l);
    const int c = k * 56 + lx + XPAD;
    const int s = ((c >> 1) * NT + nn) * 2 + (c & 1);   // pair-packed
    XsH[s] = h;
    XsL[s] = l;
  }

  // ---- stage weight tables (bf16-split, [epair][i] pair-packed) ----------
  for (int idx = tid; idx < 64 * 192; idx += 256) {
    const int i = idx / 192;
    const int e = idx - i * 192;
    const int r = e >> 6, j = e & 63;
    const int s = ((e >> 1) * WSTD + i) * 2 + (e & 1);
    __bf16 h, l;
    split_bf16(w[((i * 3 + r) * 2 + 1) * 64 + j], h, l);  // l=1 taps
    WhH[s] = h; WhL[s] = l;
    split_bf16(w[((i * 3 + r) * 2 + 0) * 64 + j], h, l);  // l=0 taps
    WlH[s] = h; WlL[s] = l;
  }
  __syncthreads();

  // ---- compute: wave -> (i-tile, 28 m rows), columns = 14 n values -------
  const int lane  = tid & 31;
  const int wid   = tid >> 5;          // 8 wave32 per block
  const int i0    = (wid & 3) * 16;
  const int mbeg  = (wid >> 2) * 28;
  const int khalf = lane >> 4;
  const int row   = lane & 15;
  const int coln  = (row < NT) ? row : (NT - 1);  // clamp dead columns
  const int aoff  = i0 + row;

  for (int m = mbeg; m < mbeg + 28; ++m) {
    v8f C = {0.f, 0.f, 0.f, 0.f, 0.f, 0.f, 0.f, 0.f};
    const int hi_ci = ((m == 0) ? 54 * 64 : (m - 2) * 64) + XPAD;  // y[(m-1)%56, l=1]
    const int lo_ci = (m - 1) * 64 + XPAD;                         // y[m, l=0]

    auto window = [&](const unsigned int* WH, const unsigned int* WL,
                      int base_ci) {
#pragma unroll
      for (int e0 = 0; e0 < 192; e0 += 32) {
        const int wp = e0 >> 1;
        const int xp = (base_ci + e0) >> 1;
        v16bf ah = ldA(WH, wp, aoff, khalf);
        v16bf al = ldA(WL, wp, aoff, khalf);
        v16bf bh = ldB(XDh, xp, coln, khalf);
        v16bf bl = ldB(XDl, xp, coln, khalf);
        // 3-term split: hi*hi + hi*lo + lo*hi  (f32 accumulate)
        C = __builtin_amdgcn_wmma_f32_16x16x32_bf16(false, ah, false, bh,
                                                    (short)0, C, false, false);
        C = __builtin_amdgcn_wmma_f32_16x16x32_bf16(false, ah, false, bl,
                                                    (short)0, C, false, false);
        C = __builtin_amdgcn_wmma_f32_16x16x32_bf16(false, al, false, bh,
                                                    (short)0, C, false, false);
      }
    };
    window(WhiH, WhiL, hi_ci);
    window(WloH, WloL, lo_ci);

    // D 16x16 f32: vgpr v, lane -> (M = v + 8*khalf, N = row). Fuse n-roll.
    if (row < NT) {
      int nout = n0 + row + 1;
      if (nout >= 56) nout -= 56;
      const size_t base =
          (((size_t)o * 64 + (size_t)(i0 + khalf * 8)) * 56 + (size_t)m) * 56 +
          (size_t)nout;
#pragma unroll
      for (int v = 0; v < 8; ++v) out[base + (size_t)v * 3136] = C[v];
    }
  }
}

extern "C" void kernel_launch(void* const* d_in, const int* in_sizes, int n_in,
                              void* d_out, int out_size, void* d_ws, size_t ws_size,
                              hipStream_t stream) {
  (void)in_sizes; (void)n_in; (void)out_size; (void)d_ws; (void)ws_size;
  const float* x = (const float*)d_in[0];
  const float* w = (const float*)d_in[1];
  float* out = (float*)d_out;

  const size_t shmem = SMEM_DW * sizeof(unsigned int);   // 312,320 B
  (void)hipFuncSetAttribute((const void*)shiftconv_wmma_bf16x3,
                            hipFuncAttributeMaxDynamicSharedMemorySize,
                            (int)shmem);

  dim3 grid(1024 * NBLK_N), block(256);
  hipLaunchKernelGGL(shiftconv_wmma_bf16x3, grid, block, shmem, stream, x, w, out);
}
